// M_70205535421338
// MI455X (gfx1250) — matrix operand order repaired and checked by reading, not compiled, not run on hardware
//
#include <hip/hip_runtime.h>
#include <hip/hip_bf16.h>
#include <math.h>

typedef __attribute__((ext_vector_type(16))) _Float16 v16h;
typedef __attribute__((ext_vector_type(4)))  _Float16 v4h;
typedef __attribute__((ext_vector_type(8)))  float    v8f;

#define M_ROWS 50176   // 256*14*14
#define K_DIM  384
#define N_DIM  1536

#define BLOCK_M 256            // 8 waves x 32 rows
#define BLOCK_N 64             // shared N panel per block
#define BS_STRIDE 392          // 384 + 8 halfs pad -> 196 dwords/row, bank-conflict-free
#define BS_STRIDE_DW (BS_STRIDE / 2)   // 196

// ---------------------------------------------------------------------------
// Kernel 1: row LayerNorm (f32 in) -> normalized f16 rows.
// One wave (32 lanes) per row; each lane owns 12 elements (3 x float4).
// ---------------------------------------------------------------------------
__global__ __launch_bounds__(256) void ln_f16_kernel(
    const float* __restrict__ x,
    const float* __restrict__ gamma,
    const float* __restrict__ beta,
    _Float16* __restrict__ xn)
{
    const int wave = threadIdx.x >> 5;
    const int lane = threadIdx.x & 31;
    const int row  = blockIdx.x * 8 + wave;

    const float4* __restrict__ xr = (const float4*)(x + (size_t)row * K_DIM);

    float4 v[3];
#pragma unroll
    for (int i = 0; i < 3; ++i)
        v[i] = xr[lane + 32 * i];

    float s = 0.f;
#pragma unroll
    for (int i = 0; i < 3; ++i) s += v[i].x + v[i].y + v[i].z + v[i].w;
#pragma unroll
    for (int off = 16; off > 0; off >>= 1) s += __shfl_xor(s, off, 32);
    const float mean = s * (1.0f / (float)K_DIM);

    float q = 0.f;
#pragma unroll
    for (int i = 0; i < 3; ++i) {
        float a = v[i].x - mean, b = v[i].y - mean,
              c = v[i].z - mean, d = v[i].w - mean;
        q += a * a + b * b + c * c + d * d;
    }
#pragma unroll
    for (int off = 16; off > 0; off >>= 1) q += __shfl_xor(q, off, 32);
    const float rstd = rsqrtf(q * (1.0f / (float)K_DIM) + 1e-6f);

    const float4* __restrict__ g4 = (const float4*)gamma;
    const float4* __restrict__ b4 = (const float4*)beta;
    _Float16* __restrict__ orow = xn + (size_t)row * K_DIM;

#pragma unroll
    for (int i = 0; i < 3; ++i) {
        const float4 gv = g4[lane + 32 * i];
        const float4 bv = b4[lane + 32 * i];
        v4h h;
        h[0] = (_Float16)((v[i].x - mean) * rstd * gv.x + bv.x);
        h[1] = (_Float16)((v[i].y - mean) * rstd * gv.y + bv.y);
        h[2] = (_Float16)((v[i].z - mean) * rstd * gv.z + bv.z);
        h[3] = (_Float16)((v[i].w - mean) * rstd * gv.w + bv.w);
        *(v4h*)(orow + lane * 4 + i * 128) = h;
    }
}

// ---------------------------------------------------------------------------
// Kernel 2: W f32 -> f16 (row-major [N=1536][K=384], layout preserved)
// ---------------------------------------------------------------------------
__global__ __launch_bounds__(256) void cvt_w_kernel(
    const float* __restrict__ W, _Float16* __restrict__ Wh, int n)
{
    int i = blockIdx.x * 256 + threadIdx.x;
    if (i < n) Wh[i] = (_Float16)W[i];
}

// ---------------------------------------------------------------------------
// Kernel 3: WMMA GEMM  out[M,N] = Xn[M,K] * W^T  (+ bias, exact GELU)
//
// Block tile: 256M x 64N. B panel (64 x 384 f16 = 48 KB) staged in LDS once
// per block (padded rows, conflict-free ds_load_b128). Each wave: 32M x 64N
// = 8 accumulators; per K-step: 2 A frags (global b128) + ALL 4 B frags
// issued up front (8 ds_load_b128) so partial dscnt waits overlap WMMAs.
//
// A frag: lane l: M = l&15, half = l>>4; 16B at K = kb+half*8, 16B at +32B.
// B frag: lane l: N = 16t + (l&15); 16B at K = kb + half*8, 16B at +32B.
// C/D:    VGPR r, lane l -> (M = r + 8*half, N = l&15)
// ---------------------------------------------------------------------------
__global__ __launch_bounds__(256) void wmma_gemm_kernel(
    const _Float16* __restrict__ Xn,
    const _Float16* __restrict__ Wh,
    const float* __restrict__ bias,
    float* __restrict__ out)
{
    __shared__ _Float16 Bs[BLOCK_N * BS_STRIDE];   // 50,176 bytes

    const int tid  = threadIdx.x;
    const int wave = tid >> 5;
    const int lane = tid & 31;

    const int mb = blockIdx.x / (N_DIM / BLOCK_N);   // 0..195
    const int ng = blockIdx.x % (N_DIM / BLOCK_N);   // 0..23
    const int n0 = ng * BLOCK_N;

    // ---- stage B panel into LDS (coalesced dword copy, once per block) ----
    {
        const uint32_t* __restrict__ wsrc =
            (const uint32_t*)(Wh + (size_t)n0 * K_DIM);      // 64 rows x 192 dw
        uint32_t* __restrict__ bdst = (uint32_t*)Bs;
#pragma unroll
        for (int it = 0; it < (BLOCK_N * (K_DIM / 2)) / 256; ++it) {
            const int idx = it * 256 + tid;                  // 0..12287
            const int n  = idx / (K_DIM / 2);
            const int kd = idx % (K_DIM / 2);
            bdst[n * BS_STRIDE_DW + kd] = wsrc[n * (K_DIM / 2) + kd];
        }
    }
    __syncthreads();

    const int half = lane >> 4;
    const int lm   = lane & 15;
    const int m0   = mb * BLOCK_M + wave * 32;

    const _Float16* __restrict__ arow0 =
        Xn + (size_t)(m0 + lm) * K_DIM + half * 8;
    const _Float16* __restrict__ arow1 = arow0 + (size_t)16 * K_DIM;
    const uint32_t* __restrict__ bs = (const uint32_t*)Bs;

    v8f acc[8] = {v8f{}, v8f{}, v8f{}, v8f{}, v8f{}, v8f{}, v8f{}, v8f{}};

#pragma unroll 2
    for (int kb = 0; kb < K_DIM; kb += 32) {
        // ---- A fragments: 4x global_load_b128 ----
        union { v16h v; uint32_t u[8]; } A0, A1;
#pragma unroll
        for (int j = 0; j < 8; ++j) {
            const int koff = (j < 4) ? (2 * j) : (8 + 2 * j); // 0,2,4,6 | 16..22
            A0.u[j] = *(const uint32_t*)(arow0 + kb + koff);
            A1.u[j] = *(const uint32_t*)(arow1 + kb + koff);
        }
        // ---- ALL B fragments issued up front: 8x ds_load_b128 ----
        union { v16h v; uint32_t u[8]; } B[4];
#pragma unroll
        for (int t = 0; t < 4; ++t) {
            const int base = (t * 16 + lm) * BS_STRIDE_DW + (kb >> 1) + half * 8;
#pragma unroll
            for (int j = 0; j < 8; ++j)
                B[t].u[j] = bs[base - half * 4 + ((j < 4) ? j : j + 4)];
        }
        // ---- 8 WMMAs; partial waits let later DS loads fly under earlier MMAs
#pragma unroll
        for (int t = 0; t < 4; ++t) {
            acc[2 * t]     = __builtin_amdgcn_wmma_f32_16x16x32_f16(
                false, A0.v, false, B[t].v, (short)0, acc[2 * t], false, false);
            acc[2 * t + 1] = __builtin_amdgcn_wmma_f32_16x16x32_f16(
                false, A1.v, false, B[t].v, (short)0, acc[2 * t + 1], false, false);
        }
    }

    // ---- epilogue: bias + exact (erf) GELU, streaming stores ----
#pragma unroll
    for (int t = 0; t < 4; ++t) {
        const int n = n0 + t * 16 + lm;
        const float bv = bias[n];
#pragma unroll
        for (int s = 0; s < 2; ++s) {
            const int mbase = m0 + s * 16 + 8 * half;
#pragma unroll
            for (int r = 0; r < 8; ++r) {
                float y = acc[2 * t + s][r] + bv;
                float g = 0.5f * y * (1.0f + erff(y * 0.70710678118654752f));
                __builtin_nontemporal_store(
                    g, &out[(size_t)(mbase + r) * N_DIM + n]);
            }
        }
    }
}

// ---------------------------------------------------------------------------
extern "C" void kernel_launch(void* const* d_in, const int* in_sizes, int n_in,
                              void* d_out, int out_size, void* d_ws, size_t ws_size,
                              hipStream_t stream)
{
    const float* x     = (const float*)d_in[0];   // (256,14,14,384) f32
    const float* gamma = (const float*)d_in[1];   // (384,)
    const float* beta  = (const float*)d_in[2];   // (384,)
    const float* W     = (const float*)d_in[3];   // (1536,384)
    const float* bias  = (const float*)d_in[4];   // (1536,)
    float* out = (float*)d_out;                   // (256,14,14,1536) f32

    _Float16* xn = (_Float16*)d_ws;                                       // 38.5 MB
    _Float16* wh = (_Float16*)((char*)d_ws + (size_t)M_ROWS * K_DIM * 2); // +1.2 MB

    ln_f16_kernel<<<M_ROWS / 8, 256, 0, stream>>>(x, gamma, beta, xn);
    cvt_w_kernel<<<(N_DIM * K_DIM + 255) / 256, 256, 0, stream>>>(W, wh, N_DIM * K_DIM);

    const int nblocks = (M_ROWS / BLOCK_M) * (N_DIM / BLOCK_N);  // 196*24 = 4704
    wmma_gemm_kernel<<<nblocks, 256, 0, stream>>>(xn, wh, bias, out);
}
